// MultiHeadCrossAttention_2190433321076
// MI455X (gfx1250) — compile-verified, matrix operand order checked by reading, not run on hardware
//
#include <hip/hip_runtime.h>
#include <hip/hip_bf16.h>
#include <math.h>

// ---------------------------------------------------------------------------
// MI455X / gfx1250 multi-head cross-attention, bf16 WMMA pipeline with
// double-buffered async global->LDS staging (ASYNCcnt) where available.
// ---------------------------------------------------------------------------

typedef __attribute__((ext_vector_type(16))) __bf16 bf16x16;
typedef __attribute__((ext_vector_type(8)))  __bf16 bf16x8;
typedef __attribute__((ext_vector_type(8)))  float  floatx8;

#define D_MODEL 1024
#define NHEADS  16
#define HDIM    64

// ---- async global->LDS support probe --------------------------------------
#if defined(__has_builtin)
#if __has_builtin(__builtin_amdgcn_global_load_async_to_lds_b128) && \
    __has_builtin(__builtin_amdgcn_s_wait_asynccnt)
#define HAVE_ASYNC_LDS 1
#endif
#endif
#ifndef HAVE_ASYNC_LDS
#define HAVE_ASYNC_LDS 0
#endif

#if HAVE_ASYNC_LDS
// Builtin signature (from hipcc diagnostic): pointers to 16-byte int vectors
// in AS1 (global) / AS3 (LDS), plus immediate offset and cpol.
typedef int v4i_gcc __attribute__((vector_size(16)));
typedef __attribute__((address_space(1))) v4i_gcc* gas128_t;
typedef __attribute__((address_space(3))) v4i_gcc* las128_t;
// 16-byte async copy global -> LDS (tracked by ASYNCcnt, no VGPR round trip)
__device__ __forceinline__ void async_cp16(const void* g, void* l) {
  __builtin_amdgcn_global_load_async_to_lds_b128((gas128_t)g, (las128_t)l, 0, 0);
}
#define WAIT_ASYNC(n) __builtin_amdgcn_s_wait_asynccnt(n)
#else
__device__ __forceinline__ void async_cp16(const void* g, void* l) {
  *(uint4*)l = *(const uint4*)g;
}
#define WAIT_ASYNC(n) ((void)0)
#endif

__device__ __forceinline__ unsigned short f32_to_bf16u(float f) {
  union { float f; unsigned int u; } v; v.f = f;
  unsigned int r = v.u + 0x7FFFu + ((v.u >> 16) & 1u);  // round-to-nearest-even
  return (unsigned short)(r >> 16);
}

// Load one 16x32 bf16 WMMA operand fragment (A layout; also the B fragment
// when the B matrix is stored transposed / column-per-row in LDS).
//   lanes 0-15 : row M=lane,    v0..3 = K 0..7,  v4..7 = K 16..23
//   lanes16-31 : row M=lane-16, v0..3 = K 8..15, v4..7 = K 24..31
__device__ __forceinline__ bf16x16 load_frag(const unsigned short* base, int stride) {
  const int lane = (int)(threadIdx.x & 31);
  const int r    = lane & 15;
  const int kb   = (lane >> 4) * 8;
  const unsigned short* p = base + r * stride;
  bf16x8 h0 = *(const bf16x8*)(p + kb);
  bf16x8 h1 = *(const bf16x8*)(p + 16 + kb);
  bf16x16 out;
#pragma unroll
  for (int i = 0; i < 8; ++i) { out[i] = h0[i]; out[i + 8] = h1[i]; }
  return out;
}

__device__ __forceinline__ floatx8 wmma_bf16(bf16x16 a, bf16x16 b, floatx8 c) {
  // (neg_a, A, neg_b, B, c_mod, C, reuse_a, reuse_b)
  return __builtin_amdgcn_wmma_f32_16x16x32_bf16(false, a, false, b, (short)0, c,
                                                 false, false);
}

// ---------------------------------------------------------------------------
// f32 -> bf16 elementwise conversion (vectorized 4-wide).
// ---------------------------------------------------------------------------
__global__ __launch_bounds__(256)
void k_convert_bf16(const float* __restrict__ in, unsigned short* __restrict__ out,
                    int n4) {
  int i = (int)blockIdx.x * 256 + (int)threadIdx.x;
  if (i < n4) {
    float4 v = ((const float4*)in)[i];
    unsigned long long p =
        (unsigned long long)f32_to_bf16u(v.x) |
        ((unsigned long long)f32_to_bf16u(v.y) << 16) |
        ((unsigned long long)f32_to_bf16u(v.z) << 32) |
        ((unsigned long long)f32_to_bf16u(v.w) << 48);
    ((unsigned long long*)out)[i] = p;
  }
}

// ---------------------------------------------------------------------------
// W[K][N] f32 -> Wt[N][K] bf16 (LDS tile transpose, bank-conflict padded).
// ---------------------------------------------------------------------------
__global__ __launch_bounds__(256)
void k_transpose_bf16(const float* __restrict__ W, unsigned short* __restrict__ Wt,
                      int K, int N) {
  __shared__ float tile[32][33];
  const int n0 = (int)blockIdx.x * 32;
  const int k0 = (int)blockIdx.y * 32;
  const int tx = (int)(threadIdx.x & 31);
  const int ty = (int)(threadIdx.x >> 5);  // 0..7
#pragma unroll
  for (int i = 0; i < 32; i += 8)
    tile[ty + i][tx] = W[(size_t)(k0 + ty + i) * N + n0 + tx];
  __syncthreads();
#pragma unroll
  for (int i = 0; i < 32; i += 8)
    Wt[(size_t)(n0 + ty + i) * K + k0 + tx] = f32_to_bf16u(tile[tx][ty + i]);
}

// ---------------------------------------------------------------------------
// C[M][N] = A[M][K](bf16) @ Wt[N][K](bf16, weight transposed) + bias (f32)
// 256 threads = 8 waves; tile 128(M) x 128(N), K-step 64, double-buffered
// LDS staged via async global->LDS copies.  Each wave owns 32x64 output,
// 16 WMMAs per K-step.
// ---------------------------------------------------------------------------
template <bool OUT_F32>
__global__ __launch_bounds__(256)
void k_gemm_bf16(const unsigned short* __restrict__ A,
                 const unsigned short* __restrict__ Wt,
                 const float* __restrict__ bias, void* __restrict__ Cout,
                 int M, int N, int K) {
  __shared__ __align__(16) unsigned short As[2][128][64];
  __shared__ __align__(16) unsigned short Bs[2][128][64];

  const int tid   = (int)threadIdx.x;
  const int wave  = tid >> 5;
  const int lane  = tid & 31;
  const int tileM = (int)blockIdx.y * 128;
  const int tileN = (int)blockIdx.x * 128;
  const int wm    = (wave & 3) * 32;
  const int wn    = (wave >> 2) * 64;

  floatx8 acc[2][4];
#pragma unroll
  for (int i = 0; i < 2; ++i)
#pragma unroll
    for (int j = 0; j < 4; ++j)
#pragma unroll
      for (int e = 0; e < 8; ++e) acc[i][j][e] = 0.0f;

  // Stage one 128x64 A tile + 128x64 Wt tile: 8 async 16B ops per thread.
  auto stage = [&](int buf, int k0) {
#pragma unroll
    for (int i = 0; i < 4; ++i) {
      int idx = tid + i * 256;
      int r = idx >> 3, sg = (idx & 7) * 8;
      async_cp16(A  + (size_t)(tileM + r) * K + k0 + sg, &As[buf][r][sg]);
      async_cp16(Wt + (size_t)(tileN + r) * K + k0 + sg, &Bs[buf][r][sg]);
    }
  };

  stage(0, 0);  // prologue: 8 async ops in flight

  const int nk = K >> 6;
  for (int kt = 0; kt < nk; ++kt) {
    const int cur = kt & 1;
    if (kt + 1 < nk) {
      stage(cur ^ 1, (kt + 1) << 6);  // 16 in flight
      WAIT_ASYNC(8);                  // in-order: current stage's 8 retired
    } else {
      WAIT_ASYNC(0);
    }
    __syncthreads();  // all threads' current-stage data visible in LDS

#pragma unroll
    for (int ks = 0; ks < 2; ++ks) {  // two 32-wide K sub-steps
      bf16x16 af[2], bfm[4];
#pragma unroll
      for (int i = 0; i < 2; ++i)
        af[i] = load_frag(&As[cur][wm + i * 16][ks * 32], 64);
#pragma unroll
      for (int j = 0; j < 4; ++j)
        bfm[j] = load_frag(&Bs[cur][wn + j * 16][ks * 32], 64);
#pragma unroll
      for (int i = 0; i < 2; ++i)
#pragma unroll
        for (int j = 0; j < 4; ++j)
          acc[i][j] = wmma_bf16(af[i], bfm[j], acc[i][j]);
    }
    __syncthreads();  // readers done: next iteration may overwrite this buffer
  }

  // Epilogue: C layout is lane(0-15)=col, reg r -> row r (lanes<16) / r+8.
  const int rofs = (lane >> 4) * 8;
  const int cl   = lane & 15;
#pragma unroll
  for (int i = 0; i < 2; ++i)
#pragma unroll
    for (int j = 0; j < 4; ++j) {
      int col  = tileN + wn + j * 16 + cl;
      float bv = bias[col];
#pragma unroll
      for (int r = 0; r < 8; ++r) {
        size_t row = (size_t)(tileM + wm + i * 16 + rofs + r);
        float v = acc[i][j][r] + bv;
        if (OUT_F32)
          ((float*)Cout)[row * N + col] = v;
        else
          ((unsigned short*)Cout)[row * N + col] = f32_to_bf16u(v);
      }
    }
}

// ---------------------------------------------------------------------------
// Flash-style attention for one (b, h, 64-row Q tile).
// 128 threads = 4 waves; each wave owns 16 Q rows.
// Q and K tiles staged via async global->LDS; V transposed through VGPRs.
// ---------------------------------------------------------------------------
__global__ __launch_bounds__(128)
void k_attention(const unsigned short* __restrict__ Qg,
                 const unsigned short* __restrict__ Kg,
                 const unsigned short* __restrict__ Vg,
                 unsigned short* __restrict__ Og, int S) {
  __shared__ __align__(16) unsigned short Qs[64][64];
  __shared__ __align__(16) unsigned short Ks[64][64];
  __shared__ __align__(16) unsigned short Vts[64][64];  // [d][kv]
  __shared__ __align__(16) unsigned short Ps[64][64];

  const int tid  = (int)threadIdx.x;
  const int wave = tid >> 5;
  const int lane = tid & 31;
  const int b    = (int)blockIdx.z;
  const int h    = (int)blockIdx.y;
  const int q0   = (int)blockIdx.x * 64;
  const int hcol = h * HDIM;

  const size_t qrow0 = (size_t)b * S + q0;
  const size_t krow0 = (size_t)b * S;

  // Stage Q tile (64x64 bf16) asynchronously; waited before first use below.
#pragma unroll
  for (int i = 0; i < 4; ++i) {
    int idx = tid + i * 128;
    int r = idx >> 3, sg = (idx & 7) * 8;
    async_cp16(Qg + (qrow0 + r) * D_MODEL + hcol + sg, &Qs[r][sg]);
  }

  float mrow[8], lrow[8];
  floatx8 Oacc[4];
#pragma unroll
  for (int r = 0; r < 8; ++r) { mrow[r] = -3.0e30f; lrow[r] = 0.0f; }
#pragma unroll
  for (int t = 0; t < 4; ++t)
#pragma unroll
    for (int e = 0; e < 8; ++e) Oacc[t][e] = 0.0f;

  const float scale = 0.125f;  // 1/sqrt(HDIM)
  const int rofs = (lane >> 4) * 8;
  const int cl   = lane & 15;

  for (int kv0 = 0; kv0 < S; kv0 += 64) {
    __syncthreads();  // prior readers of Ks/Vts/Ps done

    // K tile row-major [kv][d] via async copies; a row of K is a column of
    // K^T, so load_frag on Ks directly yields the WMMA B fragment of K^T.
#pragma unroll
    for (int i = 0; i < 4; ++i) {
      int idx = tid + i * 128;
      int r = idx >> 3, sg = (idx & 7) * 8;
      async_cp16(Kg + (krow0 + kv0 + r) * D_MODEL + hcol + sg, &Ks[r][sg]);
    }
    // V tile transposed into Vts[d][kv] so load_frag gives the B fragment.
#pragma unroll
    for (int i = 0; i < 4; ++i) {
      int idx = tid + i * 128;
      int r = idx >> 3, sg = (idx & 7) * 8;
      uint4 d4 = *(const uint4*)(Vg + (krow0 + kv0 + r) * D_MODEL + hcol + sg);
      const unsigned short* e = (const unsigned short*)&d4;
#pragma unroll
      for (int q = 0; q < 8; ++q) Vts[sg + q][r] = e[q];
    }
    WAIT_ASYNC(0);    // K (and first-iteration Q) async copies landed
    __syncthreads();

    // ---- scores: 16x64 per wave = 4 n-tiles x 2 K-steps of WMMA ----
    floatx8 sacc[4];
#pragma unroll
    for (int t = 0; t < 4; ++t)
#pragma unroll
      for (int e = 0; e < 8; ++e) sacc[t][e] = 0.0f;

    bf16x16 qf0 = load_frag(&Qs[wave * 16][0], 64);
    bf16x16 qf1 = load_frag(&Qs[wave * 16][32], 64);
#pragma unroll
    for (int t = 0; t < 4; ++t) {
      sacc[t] = wmma_bf16(qf0, load_frag(&Ks[t * 16][0], 64), sacc[t]);
      sacc[t] = wmma_bf16(qf1, load_frag(&Ks[t * 16][32], 64), sacc[t]);
    }

    // ---- online softmax: each row lives in one 16-lane half ----
#pragma unroll
    for (int r = 0; r < 8; ++r) {
      float v = fmaxf(fmaxf(sacc[0][r], sacc[1][r]),
                      fmaxf(sacc[2][r], sacc[3][r]));
#pragma unroll
      for (int msk = 1; msk < 16; msk <<= 1)
        v = fmaxf(v, __shfl_xor(v, msk, 32));
      v *= scale;
      float newm  = fmaxf(mrow[r], v);
      float fresc = __expf(mrow[r] - newm);
      mrow[r] = newm;
      float rs = 0.0f;
#pragma unroll
      for (int t = 0; t < 4; ++t) {
        float pe = __expf(sacc[t][r] * scale - newm);
        rs += pe;
        Ps[wave * 16 + rofs + r][t * 16 + cl] = f32_to_bf16u(pe);
      }
      lrow[r] = lrow[r] * fresc + rs;
#pragma unroll
      for (int t = 0; t < 4; ++t) Oacc[t][r] *= fresc;
    }
    __syncthreads();

    // ---- O += P * V ----
    bf16x16 pf0 = load_frag(&Ps[wave * 16][0], 64);
    bf16x16 pf1 = load_frag(&Ps[wave * 16][32], 64);
#pragma unroll
    for (int t = 0; t < 4; ++t) {
      Oacc[t] = wmma_bf16(pf0, load_frag(&Vts[t * 16][0], 64), Oacc[t]);
      Oacc[t] = wmma_bf16(pf1, load_frag(&Vts[t * 16][32], 64), Oacc[t]);
    }
  }

  // Normalize (lane-reduce row sums) and write concat layout [b*S+q][h*64+d].
#pragma unroll
  for (int r = 0; r < 8; ++r) {
    float lv = lrow[r];
#pragma unroll
    for (int msk = 1; msk < 16; msk <<= 1) lv += __shfl_xor(lv, msk, 32);
    float inv = 1.0f / lv;
    size_t row = qrow0 + wave * 16 + rofs + r;
#pragma unroll
    for (int t = 0; t < 4; ++t)
      Og[row * D_MODEL + hcol + t * 16 + cl] = f32_to_bf16u(Oacc[t][r] * inv);
  }
}

// ---------------------------------------------------------------------------
// Host-side orchestration.  Workspace (bf16 = ushort), total ~56 MiB:
//   Xbf, ybf : 2 x 4M | qWt..oWt : 4 x 1M | Qb,Kb,Vb : 3 x 4M | Cc : 4M
// ---------------------------------------------------------------------------
extern "C" void kernel_launch(void* const* d_in, const int* in_sizes, int n_in,
                              void* d_out, int out_size, void* d_ws,
                              size_t ws_size, hipStream_t stream) {
  (void)in_sizes; (void)n_in; (void)out_size; (void)ws_size;

  const float* X  = (const float*)d_in[0];
  const float* y  = (const float*)d_in[1];
  const float* qW = (const float*)d_in[2];
  const float* qB = (const float*)d_in[3];
  const float* kW = (const float*)d_in[4];
  const float* kB = (const float*)d_in[5];
  const float* vW = (const float*)d_in[6];
  const float* vB = (const float*)d_in[7];
  const float* oW = (const float*)d_in[8];
  const float* oB = (const float*)d_in[9];

  const int    B     = 2, S = 2048;
  const int    M     = B * S;                  // 4096 token rows
  const size_t elems = (size_t)M * D_MODEL;    // 4M
  const size_t welem = (size_t)D_MODEL * D_MODEL;

  unsigned short* p   = (unsigned short*)d_ws;
  unsigned short* Xbf = p; p += elems;
  unsigned short* ybf = p; p += elems;
  unsigned short* qWt = p; p += welem;
  unsigned short* kWt = p; p += welem;
  unsigned short* vWt = p; p += welem;
  unsigned short* oWt = p; p += welem;
  unsigned short* Qb  = p; p += elems;
  unsigned short* Kb  = p; p += elems;
  unsigned short* Vb  = p; p += elems;
  unsigned short* Cc  = p; p += elems;

  {
    int n4 = (int)(elems / 4);
    dim3 g((n4 + 255) / 256), blk(256);
    k_convert_bf16<<<g, blk, 0, stream>>>(X, Xbf, n4);
    k_convert_bf16<<<g, blk, 0, stream>>>(y, ybf, n4);
  }
  {
    dim3 g(D_MODEL / 32, D_MODEL / 32), blk(256);
    k_transpose_bf16<<<g, blk, 0, stream>>>(qW, qWt, D_MODEL, D_MODEL);
    k_transpose_bf16<<<g, blk, 0, stream>>>(kW, kWt, D_MODEL, D_MODEL);
    k_transpose_bf16<<<g, blk, 0, stream>>>(vW, vWt, D_MODEL, D_MODEL);
    k_transpose_bf16<<<g, blk, 0, stream>>>(oW, oWt, D_MODEL, D_MODEL);
  }
  {
    dim3 g(D_MODEL / 128, M / 128), blk(256);
    k_gemm_bf16<false><<<g, blk, 0, stream>>>(ybf, qWt, qB, Qb, M, D_MODEL, D_MODEL);
    k_gemm_bf16<false><<<g, blk, 0, stream>>>(Xbf, kWt, kB, Kb, M, D_MODEL, D_MODEL);
    k_gemm_bf16<false><<<g, blk, 0, stream>>>(Xbf, vWt, vB, Vb, M, D_MODEL, D_MODEL);
  }
  {
    dim3 g(S / 64, NHEADS, B), blk(128);
    k_attention<<<g, blk, 0, stream>>>(Qb, Kb, Vb, Cc, S);
  }
  {
    dim3 g(D_MODEL / 128, M / 128), blk(256);
    k_gemm_bf16<true><<<g, blk, 0, stream>>>(Cc, oWt, oB, (float*)d_out, M,
                                             D_MODEL, D_MODEL);
  }
}
